// EncoderBlock_72413148610781
// MI455X (gfx1250) — compile-verified, matrix operand order checked by reading, not run on hardware
//
#include <hip/hip_runtime.h>
#include <hip/hip_bf16.h>

// ---------- types for WMMA ----------
typedef __attribute__((ext_vector_type(16))) __bf16 v16bf;
typedef __attribute__((ext_vector_type(8)))  float  v8f;

union FragBF {
    v16bf v;
    uint4 u[2];
};

// round-to-nearest-even f32 -> bf16
static __device__ __forceinline__ unsigned int f2bf1(float f) {
    unsigned int u = __float_as_uint(f);
    return (u + 0x7FFFu + ((u >> 16) & 1u)) >> 16;
}
static __device__ __forceinline__ unsigned int pack_bf16(float lo, float hi) {
    return (f2bf1(hi) << 16) | (f2bf1(lo) & 0xFFFFu);
}
static __device__ __forceinline__ float bf2f(unsigned short h) {
    return __uint_as_float(((unsigned int)h) << 16);
}

// ---------- small utility kernels ----------
__global__ void fill_f32(float* __restrict__ p, float v, long long n) {
    long long i = (long long)blockIdx.x * 256 + threadIdx.x;
    if (i < n) p[i] = v;
}

__global__ void f32_to_bf16_pk(const float2* __restrict__ in, unsigned int* __restrict__ out,
                               long long n2) {
    long long i = (long long)blockIdx.x * 256 + threadIdx.x;
    if (i < n2) {
        float2 v = in[i];
        out[i] = pack_bf16(v.x, v.y);
    }
}

__global__ void deg_accum(const int* __restrict__ dst, float* __restrict__ deg, long long E) {
    long long i = (long long)blockIdx.x * 256 + threadIdx.x;
    if (i < E) atomicAdd(&deg[dst[i]], 1.0f);
}

// deg -> rsqrt(deg) in place (deg >= 1 always because of the self-loop)
__global__ void deg_to_dinv(float* __restrict__ deg, int n) {
    int i = blockIdx.x * 256 + threadIdx.x;
    if (i < n) deg[i] = rsqrtf(deg[i]);
}

// ---------- repack W[K][NOUT] (f32 row-major) into WMMA B-fragment layout ----------
// Bfrag[ntile][ktile][lane][j] : 8 dwords (16 bf16) per lane. Lane n<16 -> col n,
// K rows {0..7,16..23}; lane n+16 -> col n, K rows {8..15,24..31}; dword j packs
// (K_even, K_odd).
template <int K, int NOUT>
__global__ void prep_wfrag(const float* __restrict__ W, unsigned int* __restrict__ Bfrag) {
    constexpr int KT = K / 32;
    int idx = blockIdx.x * 256 + threadIdx.x;
    int total = (NOUT / 16) * KT * 32 * 8;
    if (idx >= total) return;
    int j     = idx & 7;
    int lane  = (idx >> 3) & 31;
    int rest  = idx >> 8;
    int kt    = rest % KT;
    int ntile = rest / KT;
    int n     = ntile * 16 + (lane & 15);
    int kin   = ((lane >= 16) ? 8 : 0) + ((j < 4) ? 2 * j : 16 + 2 * (j - 4));
    int kbase = kt * 32 + kin;
    float lo = W[(size_t)kbase * NOUT + n];
    float hi = W[(size_t)(kbase + 1) * NOUT + n];
    Bfrag[idx] = pack_bf16(lo, hi);
}

// ---------- WMMA GEMM: C_bf16[M][NOUT] = A_bf16[M][K] x Bfrag ----------
// Block: 256 threads = 8 waves = 4 row-strips x 2 col-halves; block tile 64 x NOUT.
// A tiles (64x32 bf16 = 4KB) are double-buffered in LDS and streamed with
// global_load_async_to_lds_b128 (ASYNCcnt), overlapping the WMMA pipeline.
template <int K, int NOUT>
__global__ __launch_bounds__(256) void gcn_gemm_wmma(
    const unsigned short* __restrict__ A,
    const unsigned int* __restrict__ Bfrag,
    unsigned short* __restrict__ C,
    int M)
{
    constexpr int KT = K / 32;
    constexpr int NTW = NOUT / 32;               // 16-col tiles per wave
    __shared__ unsigned int Atile[2][64 * 16];   // 2 x (64 rows x 32 bf16)

    const int tid      = threadIdx.x;
    const int wave     = tid >> 5;
    const int lane     = tid & 31;
    const int rowStrip = wave & 3;
    const int colHalf  = wave >> 2;
    const int blockRow0 = blockIdx.x * 64;
    const int m    = lane & 15;
    const int kSel = lane >> 4;

    // staging mapping: each thread copies one 16B chunk per K-step
    const int srow   = tid >> 2;                 // 0..63
    const int schunk = tid & 3;                  // 16B chunk within 64B row slab
    int grow_s = blockRow0 + srow;
    if (grow_s >= M) grow_s = M - 1;             // clamp: rows >= M never stored
    const unsigned short* gsrcRow = A + (size_t)grow_s * K;
    const unsigned int ldsSlot0 =
        (unsigned int)(uintptr_t)&Atile[0][srow * 16 + schunk * 4];
    const unsigned int ldsSlot1 =
        (unsigned int)(uintptr_t)&Atile[1][srow * 16 + schunk * 4];

    v8f acc[NTW];
#pragma unroll
    for (int t = 0; t < NTW; ++t)
#pragma unroll
        for (int r = 0; r < 8; ++r) acc[t][r] = 0.0f;

    // prologue: stage tile 0
    {
        unsigned long long ga = (unsigned long long)(uintptr_t)(gsrcRow + 0 * 32 + schunk * 8);
        asm volatile("global_load_async_to_lds_b128 %0, %1, off"
                     :: "v"(ldsSlot0), "v"(ga) : "memory");
    }

    for (int kt = 0; kt < KT; ++kt) {
        if (kt + 1 < KT) {
            // stage next tile into the other buffer, then wait for current one
            unsigned long long ga =
                (unsigned long long)(uintptr_t)(gsrcRow + (kt + 1) * 32 + schunk * 8);
            unsigned int slot = ((kt + 1) & 1) ? ldsSlot1 : ldsSlot0;
            asm volatile("global_load_async_to_lds_b128 %0, %1, off"
                         :: "v"(slot), "v"(ga) : "memory");
            asm volatile("s_wait_asynccnt 0x1" ::: "memory");
        } else {
            asm volatile("s_wait_asynccnt 0x0" ::: "memory");
        }
        __syncthreads();   // all waves' async copies for tile kt have landed

        const unsigned int* buf = &Atile[kt & 1][0];
        FragBF a;
        {
            const unsigned int* base = buf + (rowStrip * 16 + m) * 16 + kSel * 4;
            a.u[0] = *(const uint4*)(base);
            a.u[1] = *(const uint4*)(base + 8);
        }

#pragma unroll
        for (int t = 0; t < NTW; ++t) {
            int ntile = colHalf * NTW + t;
            const unsigned int* bp = Bfrag + ((size_t)(ntile * KT + kt) * 32 + lane) * 8;
            FragBF b;
            b.u[0] = *(const uint4*)(bp);
            b.u[1] = *(const uint4*)(bp + 4);
            acc[t] = __builtin_amdgcn_wmma_f32_16x16x32_bf16(
                false, a.v, false, b.v, (short)0, acc[t], false, false);
        }
        __syncthreads();   // done reading buf before it is restaged
    }

    // store bf16: lane covers col (lane&15); rows r + (kSel ? 8 : 0) of the strip
    int rowBase = blockRow0 + rowStrip * 16 + (kSel ? 8 : 0);
#pragma unroll
    for (int t = 0; t < NTW; ++t) {
        int n = (colHalf * NTW + t) * 16 + m;
#pragma unroll
        for (int r = 0; r < 8; ++r) {
            int grow = rowBase + r;
            if (grow < M)
                C[(size_t)grow * NOUT + n] = (unsigned short)f2bf1(acc[t][r]);
        }
    }
}

// ---------- edge scatter: one wave per edge, lane-strided feature atomics ----------
// h is bf16 so gather stream (51.2MB) + f32 accumulator (102.4MB) fit in 192MB L2.
template <int F>
__global__ __launch_bounds__(256) void scatter_edges(
    const int* __restrict__ src, const int* __restrict__ dst,
    const float* __restrict__ dinv, const unsigned short* __restrict__ h,
    float* __restrict__ out, long long E)
{
    long long e = (long long)blockIdx.x * 8 + (threadIdx.x >> 5);
    if (e >= E) return;
    int lane = threadIdx.x & 31;
    int s = src[e], d = dst[e];
    const unsigned short* hp = h + (size_t)s * F;
    __builtin_prefetch(hp + lane * 4, 0, 3);     // global_prefetch
    float norm = dinv[s] * dinv[d];
    float* op = out + (size_t)d * F;
#pragma unroll
    for (int f = lane; f < F; f += 32)
        atomicAdd(op + f, bf2f(hp[f]) * norm);
}

// ---------- fused self-loop + bias + relu (in place), optional bf16 copy out ----------
template <int F>
__global__ void finish_layer(float* __restrict__ data, const unsigned short* __restrict__ h,
                             const float* __restrict__ dinv, const float* __restrict__ b,
                             unsigned short* __restrict__ bfout, int n)
{
    long long idx = (long long)blockIdx.x * 256 + threadIdx.x;
    long long total = (long long)n * F;
    if (idx >= total) return;
    int i = (int)(idx / F);
    int f = (int)(idx % F);
    float di = dinv[i];
    float v = data[idx] + bf2f(h[idx]) * di * di + b[f];
    v = v > 0.0f ? v : 0.0f;
    data[idx] = v;
    if (bfout) bfout[idx] = (unsigned short)f2bf1(v);
}

// ---------- host launch ----------
static inline size_t align256(size_t x) { return (x + 255) & ~(size_t)255; }
static inline unsigned nblk(long long n, int per) { return (unsigned)((n + per - 1) / per); }

extern "C" void kernel_launch(void* const* d_in, const int* in_sizes, int n_in,
                              void* d_out, int out_size, void* d_ws, size_t ws_size,
                              hipStream_t stream) {
    const float* x  = (const float*)d_in[0];
    const float* W1 = (const float*)d_in[1];
    const float* b1 = (const float*)d_in[2];
    const float* W2 = (const float*)d_in[3];
    const float* b2 = (const float*)d_in[4];
    const int* eidx = (const int*)d_in[5];

    const int D_IN = 256, D_HID = 256, D_OUT = 128;
    int N = in_sizes[0] / D_IN;                 // 100000
    long long E = in_sizes[5] / 2;              // 3200000
    const int* srcI = eidx;
    const int* dstI = eidx + E;

    // workspace carve-up
    char* ws = (char*)d_ws;
    size_t off = 0;
    float* dinv = (float*)(ws + off);               off += align256((size_t)N * 4);
    unsigned int* Wf1 = (unsigned int*)(ws + off);  off += align256((size_t)(D_HID/16)*(D_IN/32)*32*8*4);
    unsigned int* Wf2 = (unsigned int*)(ws + off);  off += align256((size_t)(D_OUT/16)*(D_HID/32)*32*8*4);
    unsigned short* XBF = (unsigned short*)(ws + off); off += align256((size_t)N * D_IN * 2);   // x bf16, reused as y1 bf16
    unsigned short* H = (unsigned short*)(ws + off);   off += align256((size_t)N * D_HID * 2);  // h1 bf16, reused as h2 bf16
    float* Y = (float*)(ws + off);                  // f32 agg1 -> relu'd layer-1 output
    float* out = (float*)d_out;

    // 1) init: deg=1 (self-loop), zero agg1, repack weights, x -> bf16
    fill_f32<<<nblk(N, 256), 256, 0, stream>>>(dinv, 1.0f, N);
    fill_f32<<<nblk((long long)N * D_HID, 256), 256, 0, stream>>>(Y, 0.0f, (long long)N * D_HID);
    prep_wfrag<D_IN, D_HID><<<nblk((D_HID/16)*(D_IN/32)*32*8, 256), 256, 0, stream>>>(W1, Wf1);
    prep_wfrag<D_HID, D_OUT><<<nblk((D_OUT/16)*(D_HID/32)*32*8, 256), 256, 0, stream>>>(W2, Wf2);
    f32_to_bf16_pk<<<nblk((long long)N * D_IN / 2, 256), 256, 0, stream>>>(
        (const float2*)x, (unsigned int*)XBF, (long long)N * D_IN / 2);

    // 2) degrees -> dinv
    deg_accum<<<nblk(E, 256), 256, 0, stream>>>(dstI, dinv, E);
    deg_to_dinv<<<nblk(N, 256), 256, 0, stream>>>(dinv, N);

    // 3) layer 1: h1 = x @ W1 (async-LDS + WMMA), scatter, fused self-loop+bias+relu
    gcn_gemm_wmma<D_IN, D_HID><<<nblk(N, 64), 256, 0, stream>>>(XBF, Wf1, H, N);
    scatter_edges<D_HID><<<nblk(E, 8), 256, 0, stream>>>(srcI, dstI, dinv, H, Y, E);
    finish_layer<D_HID><<<nblk((long long)N * D_HID, 256), 256, 0, stream>>>(Y, H, dinv, b1, XBF, N);

    // 4) layer 2: h2 = y1 @ W2, scatter directly into d_out, finish in place
    fill_f32<<<nblk((long long)N * D_OUT, 256), 256, 0, stream>>>(out, 0.0f, (long long)N * D_OUT);
    gcn_gemm_wmma<D_HID, D_OUT><<<nblk(N, 64), 256, 0, stream>>>(XBF, Wf2, H, N);
    scatter_edges<D_OUT><<<nblk(E, 8), 256, 0, stream>>>(srcI, dstI, dinv, H, out, E);
    finish_layer<D_OUT><<<nblk((long long)N * D_OUT, 256), 256, 0, stream>>>(out, H, dinv, b2,
                                                                             (unsigned short*)nullptr, N);
}
